// W8A8LinearStatic_78331613545181
// MI455X (gfx1250) — 8-bit matrix operand order repaired and checked by reading, compiled, not run on hardware
//
#include <hip/hip_runtime.h>
#include <stdint.h>

typedef __attribute__((ext_vector_type(8))) int   v8i;
typedef __attribute__((ext_vector_type(4))) int   v4i;
typedef __attribute__((ext_vector_type(2))) int   v2i;
typedef __attribute__((ext_vector_type(4))) float v4f;

typedef __attribute__((address_space(1))) v4i g_v4i;
typedef __attribute__((address_space(3))) v4i l_v4i;

__device__ __forceinline__ void async_copy16(const void* g, void* l) {
#if __has_builtin(__builtin_amdgcn_global_load_async_to_lds_b128)
  __builtin_amdgcn_global_load_async_to_lds_b128(
      (g_v4i*)(uintptr_t)g,
      (l_v4i*)(uintptr_t)(uint32_t)(uintptr_t)l,
       0,  0);
#else
  *(v4i*)l = *(const v4i*)g;
#endif
}

__device__ __forceinline__ void wait_async_le8() {
  asm volatile("s_wait_asynccnt 0x8" ::: "memory");
}
__device__ __forceinline__ void wait_async_0() {
  asm volatile("s_wait_asynccnt 0x0" ::: "memory");
}

__global__ __launch_bounds__(256)
void quant_static_kernel(const float* __restrict__ x,
                         int8_t* __restrict__ xq,
                         const float* __restrict__ scale_p,
                         const int8_t* __restrict__ off_p,
                         long long total16) {
  long long i = (long long)blockIdx.x * blockDim.x + threadIdx.x;
  if (i >= total16) return;

  const float inv = 1.0f / scale_p[0];
  const float off = (float)off_p[0];

  const v4f* x4 = (const v4f*)(x) + i * 4;
  int packed[4];
#pragma unroll
  for (int g = 0; g < 4; ++g) {
    v4f v = __builtin_nontemporal_load(&x4[g]);
    int q[4];
#pragma unroll
    for (int j = 0; j < 4; ++j) {
      float f = rintf(v[j] * inv) + off;
      f = fminf(127.0f, fmaxf(-128.0f, f));
      q[j] = (int)f;
    }
    packed[g] = (q[0] & 0xff) | ((q[1] & 0xff) << 8) |
                ((q[2] & 0xff) << 16) | ((q[3] & 0xff) << 24);
  }
  v4i outv = {packed[0], packed[1], packed[2], packed[3]};
  *((v4i*)(xq) + i) = outv;
}

#define BM 128
#define BN 128
#define BKB 128
#define LDA 144
#define LDB 144

__global__ __launch_bounds__(256)
void w8a8_wmma_gemm(const int8_t* __restrict__ xq,
                    const int8_t* __restrict__ w,
                    const int*    __restrict__ qbias,
                    const float*  __restrict__ deq,
                    float* __restrict__ out,
                    int M, int N, int K) {
  __shared__ __align__(16) int8_t As[3][BM * LDA];
  __shared__ __align__(16) int8_t Bs[3][BN * LDB];

  const int tid     = threadIdx.x;
  const int lane    = tid & 31;
  const int wid     = tid >> 5;
  const int wm      = wid & 3;
  const int wn      = wid >> 2;
  const int laneLow = lane & 15;
  const int laneHi  = lane >> 4;

  const int m0 = blockIdx.y * BM;
  const int n0 = blockIdx.x * BN;

  const int lrow = tid >> 3;
  const int lcol = (tid & 7) * 16;

  auto issue_tile = [&](int buf, int kt) {
#pragma unroll
    for (int p = 0; p < 4; ++p) {
      const int r = lrow + 32 * p;
      async_copy16(xq + (size_t)(m0 + r) * K + kt + lcol,
                   &As[buf][r * LDA + lcol]);
      async_copy16(w + (size_t)(n0 + r) * K + kt + lcol,
                   &Bs[buf][r * LDB + lcol]);
    }
  };

  const int T = K / BKB;
  issue_tile(0, 0);
  issue_tile(1, BKB);

  v8i acc[2][4] = {};

  int buf = 0, ibuf = 2;
  for (int t = 0; t < T; ++t) {
    if (t + 1 < T) wait_async_le8(); else wait_async_0();
    __syncthreads();

    if (t + 2 < T) issue_tile(ibuf, (t + 2) * BKB);

#pragma unroll
    for (int ks = 0; ks < 2; ++ks) {
      v8i afrag[2], bfrag[4];
#pragma unroll
      for (int mt = 0; mt < 2; ++mt) {
        const int8_t* base = &As[buf][(wm * 32 + mt * 16 + laneLow) * LDA +
                                      ks * 64 + laneHi * 16];
        v4i a0 = *(const v4i*)(base + 0);
        v4i a1 = *(const v4i*)(base + 32);
        afrag[mt] = (v8i){a0.x, a0.y, a0.z, a0.w, a1.x, a1.y, a1.z, a1.w};
      }
#pragma unroll
      for (int nt = 0; nt < 4; ++nt) {
        const int8_t* base = &Bs[buf][(wn * 64 + nt * 16 + laneLow) * LDB +
                                      ks * 64 + laneHi * 16];
        v4i b0 = *(const v4i*)(base + 0);
        v4i b1 = *(const v4i*)(base + 32);
        bfrag[nt] = (v8i){b0.x, b0.y, b0.z, b0.w, b1.x, b1.y, b1.z, b1.w};
      }
#pragma unroll
      for (int nt = 0; nt < 4; ++nt) {
#pragma unroll
        for (int mt = 0; mt < 2; ++mt) {
          acc[mt][nt] = __builtin_amdgcn_wmma_i32_16x16x64_iu8(
              true, afrag[mt], true, bfrag[nt], acc[mt][nt],
               false,  false);
        }
      }
    }

    if (++buf == 3) buf = 0;
    if (++ibuf == 3) ibuf = 0;
  }

#pragma unroll
  for (int nt = 0; nt < 4; ++nt) {
    const int n  = n0 + wn * 64 + nt * 16 + laneLow;
    const int qb = qbias[n];
    const float dq = deq[n];
#pragma unroll
    for (int mt = 0; mt < 2; ++mt) {
      const int rbase = m0 + wm * 32 + mt * 16 + laneHi * 8;
#pragma unroll
      for (int r = 0; r < 8; ++r) {
        float v = (float)(acc[mt][nt][r] + qb) * dq;
        __builtin_nontemporal_store(v, &out[(size_t)(rbase + r) * N + n]);
      }
    }
  }
}

extern "C" void kernel_launch(void* const* d_in, const int* in_sizes, int n_in,
                              void* d_out, int out_size, void* d_ws, size_t ws_size,
                              hipStream_t stream) {
  const float*  x      = (const float*)d_in[0];
  const int8_t* weight = (const int8_t*)d_in[1];
  const float*  iscale = (const float*)d_in[2];
  const int8_t* ioff   = (const int8_t*)d_in[3];
  const float*  deq    = (const float*)d_in[4];
  const int*    qbias  = (const int*)d_in[5];
  float* out = (float*)d_out;

  const int N = in_sizes[5];
  const int K = in_sizes[1] / N;
  const int M = in_sizes[0] / K;

  int8_t* xq = (int8_t*)d_ws;

  long long total16 = ((long long)M * K) / 16;
  dim3 qgrid((unsigned)((total16 + 255) / 256));
  quant_static_kernel<<<qgrid, 256, 0, stream>>>(x, xq, iscale, ioff, total16);

  dim3 grid(N / BN, M / BM);
  w8a8_wmma_gemm<<<grid, 256, 0, stream>>>(xq, weight, qbias, deq, out, M, N, K);
}
